// Attention_63471026700747
// MI455X (gfx1250) — compile-verified, matrix operand order checked by reading, not run on hardware
//
#include <hip/hip_runtime.h>
#include <hip/hip_bf16.h>

typedef __attribute__((ext_vector_type(16))) _Float16 v16h;
typedef __attribute__((ext_vector_type(8)))  _Float16 v8h;
typedef __attribute__((ext_vector_type(8)))  float    v8f;

// Problem constants (from reference): B=8, N=1024, C=768, H=12, hd=64
#define NB   8
#define NTOK 1024
#define CDIM 768
#define NH   12
#define HD   64
#define BHT  (NB * NH)          // 96
#define MROW (NB * NTOK)        // 8192

// ---------------------------------------------------------------------------
// CDNA5 async global->LDS copy (ASYNCcnt-tracked). 16 bytes per lane.
// LDS destination offset = low 32 bits of the generic shared pointer.
// ---------------------------------------------------------------------------
__device__ __forceinline__ unsigned lds_off(const void* p) {
  return (unsigned)(size_t)p;
}
__device__ __forceinline__ void async_copy16B(unsigned dst_lds, const void* src) {
  asm volatile("global_load_async_to_lds_b128 %0, %1, off"
               :: "v"(dst_lds), "v"(src)
               : "memory");
}
__device__ __forceinline__ void wait_async0() {
  asm volatile("s_wait_asynccnt 0x0" ::: "memory");
}

// ---------------------------------------------------------------------------
// WMMA helpers (CDNA5 gfx1250, wave32).
// 16-bit A fragment, 16x32 (MxK): lane L holds row M=L&15; half-wave sel
// hs=L>>4 picks K groups: elems 0..7 -> K = hs*8+0..7, elems 8..15 -> K =
// 16+hs*8+0..7. Both contiguous 8-half (16B) runs -> two ds_load_b128.
// B fragment for an NT GEMM has the identical index map with M<->N.
// ---------------------------------------------------------------------------
__device__ __forceinline__ v16h frag_k(const _Float16* p, int ld, int row, int lane) {
  const int hs = (lane >> 4) & 1;
  const _Float16* q = p + (size_t)(row + (lane & 15)) * ld + hs * 8;
  v8h lo = *(const v8h*)q;          // K = hs*8 + 0..7
  v8h hi = *(const v8h*)(q + 16);   // K = 16 + hs*8 + 0..7
  v16h f;
#pragma unroll
  for (int i = 0; i < 8; ++i) { f[i] = lo[i]; f[i + 8] = hi[i]; }
  return f;
}

// B fragment for an NN GEMM: operand stored [K][Ncol] in LDS (ld = row pitch).
__device__ __forceinline__ v16h frag_nn(const _Float16* p, int ld, int col, int lane) {
  const int hs = (lane >> 4) & 1;
  const int n = col + (lane & 15);
  v16h f;
#pragma unroll
  for (int i = 0; i < 16; ++i) {
    const int k = ((i & 8) << 1) + hs * 8 + (i & 7);
    f[i] = p[(size_t)k * ld + n];
  }
  return f;
}

__device__ __forceinline__ v8f wmma16(v16h a, v16h b, v8f c) {
  // (neg_a, A, neg_b, B, c_mod, C, reuse_a, reuse_b)
  return __builtin_amdgcn_wmma_f32_16x16x32_f16(false, a, false, b, (short)0, c,
                                                false, false);
}

// ---------------------------------------------------------------------------
// f32 -> f16 conversion (grid-stride)
// ---------------------------------------------------------------------------
__global__ __launch_bounds__(256) void f32_to_f16(const float* __restrict__ src,
                                                  _Float16* __restrict__ dst, int n) {
  int i = blockIdx.x * blockDim.x + threadIdx.x;
  const int stride = gridDim.x * blockDim.x;
  for (; i < n; i += stride) dst[i] = (_Float16)src[i];
}

// ---------------------------------------------------------------------------
// Tiled NT GEMM: C[M,N] = A[M,K] * W[N,K]^T, f16 in / f32 accum.
// Block tile 128x64, BK=32, 256 threads = 8 waves (4 along M x 2 along N),
// each wave owns a 32x32 patch = 2x2 WMMA tiles.
// Double-buffered LDS filled by async global->LDS copies: the next K slab is
// in flight (ASYNCcnt) while WMMAs consume the current one.
// EPI==0: QKV epilogue -> scatter into q/k/v [B,H,N,64] f16, q scaled 0.125
// EPI==1: projection epilogue -> f32 out + bias
// ---------------------------------------------------------------------------
template <int EPI>
__global__ __launch_bounds__(256) void gemm_nt(
    const _Float16* __restrict__ A, const _Float16* __restrict__ W, int K,
    _Float16* __restrict__ q_out, _Float16* __restrict__ k_out,
    _Float16* __restrict__ v_out, float* __restrict__ outp,
    const float* __restrict__ bias) {
  __shared__ __align__(16) _Float16 As[2][128 * 48];
  __shared__ __align__(16) _Float16 Bs[2][64 * 48];

  const int tid  = threadIdx.x;
  const int lane = tid & 31;
  const int wave = tid >> 5;
  const int m0 = (wave & 3) * 32;   // wave patch origin in tile
  const int n0 = (wave >> 2) * 32;
  const size_t rowA = (size_t)blockIdx.y * 128;
  const size_t rowB = (size_t)blockIdx.x * 64;

  const int arow = tid >> 1, ac0 = (tid & 1) * 16;  // A stage: 128 rows x 32
  const int brow = tid >> 2, bc0 = (tid & 3) * 8;   // B stage: 64 rows x 32

  const _Float16* ag = A + (rowA + arow) * K + ac0;  // + kb
  const _Float16* bg = W + (rowB + brow) * K + bc0;  // + kb
  const unsigned aL[2] = {lds_off(&As[0][arow * 48 + ac0]),
                          lds_off(&As[1][arow * 48 + ac0])};
  const unsigned bL[2] = {lds_off(&Bs[0][brow * 48 + bc0]),
                          lds_off(&Bs[1][brow * 48 + bc0])};

  v8f acc[2][2] = {};

  // prologue: stage slab 0 into buffer 0
  async_copy16B(aL[0], ag);
  async_copy16B(aL[0] + 16, ag + 8);
  async_copy16B(bL[0], bg);
  wait_async0();
  __syncthreads();

  int cur = 0;
  for (int kb = 0; kb < K; kb += 32) {
    const bool more = (kb + 32) < K;
    if (more) {  // kick off next slab while we compute on this one
      const int nxt = cur ^ 1;
      async_copy16B(aL[nxt], ag + kb + 32);
      async_copy16B(aL[nxt] + 16, ag + kb + 40);
      async_copy16B(bL[nxt], bg + kb + 32);
    }

    const _Float16* Ab = As[cur];
    const _Float16* Bb = Bs[cur];
    v16h af0 = frag_k(Ab, 48, m0, lane);
    v16h af1 = frag_k(Ab, 48, m0 + 16, lane);
    v16h bf0 = frag_k(Bb, 48, n0, lane);
    v16h bf1 = frag_k(Bb, 48, n0 + 16, lane);
    acc[0][0] = wmma16(af0, bf0, acc[0][0]);
    acc[0][1] = wmma16(af0, bf1, acc[0][1]);
    acc[1][0] = wmma16(af1, bf0, acc[1][0]);
    acc[1][1] = wmma16(af1, bf1, acc[1][1]);

    if (more) wait_async0();  // own async copies done; barrier covers the rest
    __syncthreads();
    cur ^= 1;
  }

#pragma unroll
  for (int tm = 0; tm < 2; ++tm)
#pragma unroll
    for (int tn = 0; tn < 2; ++tn)
#pragma unroll
      for (int v = 0; v < 8; ++v) {
        const int m = (int)rowA + m0 + tm * 16 + v + ((lane >> 4) << 3);
        const int n = (int)rowB + n0 + tn * 16 + (lane & 15);
        float val = acc[tm][tn][v];
        if constexpr (EPI == 0) {
          // n in [0,3C): split (t, h, d); m -> (batch, token)
          const int bb = m >> 10, tok = m & 1023;
          const int t = n / CDIM;
          const int rr = n - t * CDIM;
          const int hh = rr >> 6, dd = rr & 63;
          _Float16* dst = (t == 0) ? q_out : (t == 1) ? k_out : v_out;
          if (t == 0) val *= 0.125f;  // hd^-0.5
          dst[(((size_t)(bb * NH + hh)) * NTOK + tok) * HD + dd] = (_Float16)val;
        } else {
          outp[(size_t)m * CDIM + n] = val + bias[n];
        }
      }
}

// ---------------------------------------------------------------------------
// Raw attention scores: S = q * k^T per (b,h). One block computes a 64x64
// S tile. q is pre-scaled. Pure NT WMMA, K = hd = 64 (2 chained WMMAs).
// q/k blocks staged by async global->LDS copies.
// grid = (16 colblk, 16 rowblk, 96 bh)
// ---------------------------------------------------------------------------
__global__ __launch_bounds__(256) void attn_scores(
    const _Float16* __restrict__ qh, const _Float16* __restrict__ kh,
    float* __restrict__ attn) {
  __shared__ __align__(16) _Float16 qs[64 * 72];
  __shared__ __align__(16) _Float16 ks[64 * 72];
  const int tid = threadIdx.x, lane = tid & 31, wave = tid >> 5;
  const int cb = blockIdx.x, rb = blockIdx.y, bh = blockIdx.z;

  const _Float16* qb = qh + (size_t)bh * NTOK * HD + (size_t)rb * 64 * HD;
  const _Float16* kb = kh + (size_t)bh * NTOK * HD + (size_t)cb * 64 * HD;

  {  // stage both 64x64 f16 blocks: 4 async 16B copies per thread
    const int row = tid >> 2, c0 = (tid & 3) * 16;
    const unsigned qd = lds_off(qs + row * 72 + c0);
    const unsigned kd = lds_off(ks + row * 72 + c0);
    async_copy16B(qd, qb + row * HD + c0);
    async_copy16B(qd + 16, qb + row * HD + c0 + 8);
    async_copy16B(kd, kb + row * HD + c0);
    async_copy16B(kd + 16, kb + row * HD + c0 + 8);
  }
  wait_async0();
  __syncthreads();

  const int r = (wave & 3) * 16;
  const v16h a0 = frag_k(qs, 72, r, lane);        // K = 0..31
  const v16h a1 = frag_k(qs + 32, 72, r, lane);   // K = 32..63
  float* Srow = attn + (size_t)bh * NTOK * NTOK + (size_t)(rb * 64) * NTOK + cb * 64;

#pragma unroll
  for (int cc = 0; cc < 2; ++cc) {
    const int c = ((wave >> 2) * 2 + cc) * 16;
    v16h b0 = frag_k(ks, 72, c, lane);
    v16h b1 = frag_k(ks + 32, 72, c, lane);
    v8f acc = {};
    acc = wmma16(a0, b0, acc);
    acc = wmma16(a1, b1, acc);
#pragma unroll
    for (int v = 0; v < 8; ++v) {
      const int m = r + v + ((lane >> 4) << 3);
      const int n = c + (lane & 15);
      Srow[(size_t)m * NTOK + n] = acc[v];
    }
  }
}

// ---------------------------------------------------------------------------
// Row softmax over N=1024, in place on the attn output (exact f32).
// One block per row; thread t owns elements t, t+256, t+512, t+768.
// ---------------------------------------------------------------------------
__global__ __launch_bounds__(256) void softmax_rows(float* __restrict__ attn) {
  __shared__ float red[256];
  const int tid = threadIdx.x;
  float* p = attn + (size_t)blockIdx.x * NTOK;
  float v[4];
#pragma unroll
  for (int i = 0; i < 4; ++i) v[i] = p[tid + i * 256];
  float m = fmaxf(fmaxf(v[0], v[1]), fmaxf(v[2], v[3]));
  red[tid] = m;
  __syncthreads();
  for (int s = 128; s > 0; s >>= 1) {
    if (tid < s) red[tid] = fmaxf(red[tid], red[tid + s]);
    __syncthreads();
  }
  m = red[0];
  __syncthreads();
  float sum = 0.f;
#pragma unroll
  for (int i = 0; i < 4; ++i) {
    v[i] = __expf(v[i] - m);
    sum += v[i];
  }
  red[tid] = sum;
  __syncthreads();
  for (int s = 128; s > 0; s >>= 1) {
    if (tid < s) red[tid] += red[tid + s];
    __syncthreads();
  }
  const float inv = 1.0f / red[0];
#pragma unroll
  for (int i = 0; i < 4; ++i) p[tid + i * 256] = v[i] * inv;
}

// ---------------------------------------------------------------------------
// O = P * V per (b,h): NN WMMA. One block: 64 rows x 64 (= hd) cols,
// looping 16 column blocks of P. P read back f32 from d_out and converted to
// f16 through VALU (overlapping the in-flight async V copy). Result stored
// f16 into ctx laid out [B, N, C] for the projection GEMM.
// grid = (16 rowblk, 96 bh)
// ---------------------------------------------------------------------------
__global__ __launch_bounds__(256) void attn_pv(const float* __restrict__ attn,
                                               const _Float16* __restrict__ vh,
                                               _Float16* __restrict__ ctx) {
  __shared__ __align__(16) _Float16 Ps[64 * 72];
  __shared__ __align__(16) _Float16 Vs[64 * 72];
  const int tid = threadIdx.x, lane = tid & 31, wave = tid >> 5;
  const int rb = blockIdx.x, bh = blockIdx.y;
  const int b = bh / NH, h = bh - b * NH;

  const float* Pbase = attn + (size_t)bh * NTOK * NTOK + (size_t)(rb * 64) * NTOK;
  const _Float16* Vbase = vh + (size_t)bh * NTOK * HD;

  const int prow = tid >> 2, pc0 = (tid & 3) * 16;
  const unsigned vd = lds_off(Vs + prow * 72 + pc0);
  const int r = (wave & 3) * 16;
  v8f acc[2] = {};

  for (int j = 0; j < 16; ++j) {
    {  // async V block copy first, then overlap with P f32->f16 conversion
      const _Float16* vsrc = Vbase + (size_t)(j * 64 + prow) * HD + pc0;
      async_copy16B(vd, vsrc);
      async_copy16B(vd + 16, vsrc + 8);

      const float* src = Pbase + (size_t)prow * NTOK + j * 64 + pc0;
      v8h p0, p1;
#pragma unroll
      for (int i = 0; i < 8; ++i) {
        p0[i] = (_Float16)src[i];
        p1[i] = (_Float16)src[8 + i];
      }
      *(v8h*)(Ps + prow * 72 + pc0) = p0;
      *(v8h*)(Ps + prow * 72 + pc0 + 8) = p1;
    }
    wait_async0();
    __syncthreads();

    v16h a0 = frag_k(Ps, 72, r, lane);       // K(n) = 0..31 of this block
    v16h a1 = frag_k(Ps + 32, 72, r, lane);  // K(n) = 32..63
#pragma unroll
    for (int cc = 0; cc < 2; ++cc) {
      const int c = ((wave >> 2) * 2 + cc) * 16;
      v16h b0 = frag_nn(Vs, 72, c, lane);
      v16h b1 = frag_nn(Vs + 32 * 72, 72, c, lane);
      acc[cc] = wmma16(a0, b0, acc[cc]);
      acc[cc] = wmma16(a1, b1, acc[cc]);
    }
    __syncthreads();
  }

#pragma unroll
  for (int cc = 0; cc < 2; ++cc) {
    const int c = ((wave >> 2) * 2 + cc) * 16;
#pragma unroll
    for (int v = 0; v < 8; ++v) {
      const int m = r + v + ((lane >> 4) << 3);
      const int tok = rb * 64 + m;
      const int d = c + (lane & 15);
      ctx[((size_t)(b * NTOK + tok)) * CDIM + h * HD + d] = (_Float16)acc[cc][v];
    }
  }
}

// ---------------------------------------------------------------------------
extern "C" void kernel_launch(void* const* d_in, const int* in_sizes, int n_in,
                              void* d_out, int out_size, void* d_ws, size_t ws_size,
                              hipStream_t stream) {
  (void)in_sizes; (void)n_in; (void)out_size; (void)ws_size;
  const float* x      = (const float*)d_in[0];   // [8,1024,768]
  const float* w_qkv  = (const float*)d_in[1];   // [2304,768]
  const float* w_proj = (const float*)d_in[2];   // [768,768]
  const float* b_proj = (const float*)d_in[3];   // [768]

  float* outp = (float*)d_out;                          // [8,1024,768]
  float* attn = outp + (size_t)NB * NTOK * CDIM;        // [8,12,1024,1024]

  char* ws = (char*)d_ws;
  _Float16* xh  = (_Float16*)ws; ws += (size_t)MROW * CDIM * 2;        // 12.6 MB
  _Float16* wqh = (_Float16*)ws; ws += (size_t)3 * CDIM * CDIM * 2;    //  3.5 MB
  _Float16* wph = (_Float16*)ws; ws += (size_t)CDIM * CDIM * 2;        //  1.2 MB
  _Float16* qh  = (_Float16*)ws; ws += (size_t)BHT * NTOK * HD * 2;    // 12.6 MB
  _Float16* kh  = (_Float16*)ws; ws += (size_t)BHT * NTOK * HD * 2;
  _Float16* vh  = (_Float16*)ws; ws += (size_t)BHT * NTOK * HD * 2;
  _Float16* ctx = (_Float16*)ws; ws += (size_t)MROW * CDIM * 2;        // 12.6 MB

  f32_to_f16<<<1024, 256, 0, stream>>>(x, xh, MROW * CDIM);
  f32_to_f16<<<1024, 256, 0, stream>>>(w_qkv, wqh, 3 * CDIM * CDIM);
  f32_to_f16<<<576, 256, 0, stream>>>(w_proj, wph, CDIM * CDIM);

  // QKV projection: M=8192, N=2304, K=768
  gemm_nt<0><<<dim3(3 * CDIM / 64, MROW / 128), 256, 0, stream>>>(
      xh, wqh, CDIM, qh, kh, vh, nullptr, nullptr);

  // S = q k^T  (raw scores into d_out attn region)
  attn_scores<<<dim3(NTOK / 64, NTOK / 64, BHT), 256, 0, stream>>>(qh, kh, attn);

  // softmax rows in place (this IS the attn output)
  softmax_rows<<<dim3(BHT * NTOK), 256, 0, stream>>>(attn);

  // O = P V  -> ctx [B,N,C] f16
  attn_pv<<<dim3(NTOK / 64, BHT), 256, 0, stream>>>(attn, vh, ctx);

  // output projection + bias: M=8192, N=768, K=768
  gemm_nt<1><<<dim3(CDIM / 64, MROW / 128), 256, 0, stream>>>(
      ctx, wph, CDIM, nullptr, nullptr, nullptr, outp, b_proj);
}
